// MultiHeadAttention_62551903699432
// MI455X (gfx1250) — matrix operand order repaired and checked by reading, compiled, not run on hardware
//
#include <hip/hip_runtime.h>

typedef __attribute__((ext_vector_type(16))) __bf16 v16bf;
typedef __attribute__((ext_vector_type(8)))  float  v8f;
typedef __attribute__((ext_vector_type(4)))  unsigned int u32x4;
typedef __attribute__((ext_vector_type(8)))  int i32x8;
typedef __attribute__((ext_vector_type(4)))  int i32x4;

#define B_  4
#define S_  1024
#define E_  1024
#define H_  16
#define HD_ 64

union Frag { v16bf v; uint4 q[2]; };

__device__ __forceinline__ unsigned short f2bf(float f) {
  unsigned u = __float_as_uint(f);
  u += 0x7FFFu + ((u >> 16) & 1u);       // round-to-nearest-even
  return (unsigned short)(u >> 16);
}

// Element index inside a 16x32 A-fragment tile (512 bf16) for (row r 0..15, col kk 0..31)
__device__ __forceinline__ int a_elem(int r, int kk) {
  int lane = r + ((kk >> 4) << 4);
  int j    = (kk & 15) >> 1;
  return lane * 16 + j * 2 + (kk & 1);
}

// Element index inside a 32x16 B-fragment tile (512 bf16) for (kk 0..31, col nn 0..15)
__device__ __forceinline__ int b_elem(int kk, int nn) {
  int lane = nn + ((kk >> 4) << 4);
  int j    = (kk & 15) >> 1;
  return lane * 16 + j * 2 + (kk & 1);
}

// ---- TDM: 1D copy of n8 x 8-byte elements, global -> LDS (D# per ISA 08 §8) ----
__device__ __forceinline__ void tdm_load_1d(unsigned lds_off, unsigned long long gaddr, int n8) {
  u32x4 g0; i32x8 g1; i32x4 g2; i32x4 g3; i32x8 g4;
  g0[0] = 1u;                                           // count=1, user mode
  g0[1] = lds_off;                                      // lds_addr
  g0[2] = (unsigned)gaddr;                              // global_addr[31:0]
  g0[3] = (unsigned)((gaddr >> 32) & 0x1FFFFFFull) | (2u << 30);  // addr[56:32], type=2
  g1[0] = 3 << 16;                                      // data_size=8B
  g1[1] = (n8 & 0xFFFF) << 16;                          // tensor_dim0[15:0]
  g1[2] = ((n8 >> 16) & 0xFFFF) | (1 << 16);            // tensor_dim0[31:16], tensor_dim1=1
  g1[3] = (n8 & 0xFFFF) << 16;                          // tile_dim0 = n8
  g1[4] = 0;                                            // tile_dim1=0 (1D), tile_dim2=0
  g1[5] = n8;                                           // tensor_dim0_stride
  g1[6] = 0; g1[7] = 0;
  g2[0] = 1; g2[1] = 0; g2[2] = 0; g2[3] = 0;           // tensor_dim2=1, rest unused
  g3[0] = 0; g3[1] = 0; g3[2] = 0; g3[3] = 0;
  g4[0] = 0; g4[1] = 0; g4[2] = 0; g4[3] = 0; g4[4] = 0; g4[5] = 0; g4[6] = 0; g4[7] = 0;
  __builtin_amdgcn_tensor_load_to_lds(g0, g1, g2, g3, g4, 0);
}

// ---- TDM: 2D copy rows x n8 (8B elems), row stride (8B units), global -> LDS ----
__device__ __forceinline__ void tdm_load_2d(unsigned lds_off, unsigned long long gaddr,
                                            int n8, int rows, int stride8) {
  u32x4 g0; i32x8 g1; i32x4 g2; i32x4 g3; i32x8 g4;
  g0[0] = 1u;
  g0[1] = lds_off;
  g0[2] = (unsigned)gaddr;
  g0[3] = (unsigned)((gaddr >> 32) & 0x1FFFFFFull) | (2u << 30);
  g1[0] = 3 << 16;                                      // data_size=8B
  g1[1] = (n8 & 0xFFFF) << 16;                          // tensor_dim0[15:0]
  g1[2] = ((n8 >> 16) & 0xFFFF) | ((rows & 0xFFFF) << 16);  // tensor_dim1 = rows
  g1[3] = (n8 & 0xFFFF) << 16;                          // tile_dim0 = n8
  g1[4] = rows & 0xFFFF;                                // tile_dim1 = rows, tile_dim2=0
  g1[5] = stride8;                                      // tensor_dim0_stride
  g1[6] = 0; g1[7] = 0;
  g2[0] = 1; g2[1] = 0; g2[2] = 0; g2[3] = 0;
  g3[0] = 0; g3[1] = 0; g3[2] = 0; g3[3] = 0;
  g4[0] = 0; g4[1] = 0; g4[2] = 0; g4[3] = 0; g4[4] = 0; g4[5] = 0; g4[6] = 0; g4[7] = 0;
  __builtin_amdgcn_tensor_load_to_lds(g0, g1, g2, g3, g4, 0);
}

// ---- Pack row-major f32 [Mrows][KT*32] into A-swizzled bf16 tiles (tile = mt*KT+kt) ----
__global__ void pack_a_f32(const float* __restrict__ src, unsigned short* __restrict__ dst, int KT) {
  int idx  = blockIdx.x * blockDim.x + threadIdx.x;
  int tile = idx >> 9, e = idx & 511;
  int mt = tile / KT, kt = tile % KT;
  int lane = e >> 4, p = e & 15, j = p >> 1, half = p & 1;
  int r = lane & 15, khi = lane >> 4;
  int kk  = (khi << 4) + 2 * j + half;
  int m = mt * 16 + r, k = kt * 32 + kk;
  dst[idx] = f2bf(src[m * (KT * 32) + k]);
}

// ---- Pack fused Wqkv: B-swizzled bf16, K=1024, N=3072, tile = nt*32+kt ----
__global__ void pack_wqkv(const float* __restrict__ Wq, const float* __restrict__ Wk,
                          const float* __restrict__ Wv, unsigned short* __restrict__ dst) {
  int idx  = blockIdx.x * blockDim.x + threadIdx.x;
  int tile = idx >> 9, e = idx & 511;
  int nt = tile >> 5, kt = tile & 31;
  int lane = e >> 4, p = e & 15, j = p >> 1, half = p & 1;
  int nn = lane & 15, khi = lane >> 4;
  int kk = (khi << 4) + 2 * j + half;
  int n = nt * 16 + nn, k = kt * 32 + kk;
  int which = n >> 10, ecol = n & 1023;
  int h = ecol >> 6, d = ecol & 63;
  const float* W = (which == 0) ? Wq : ((which == 1) ? Wk : Wv);
  dst[idx] = f2bf(W[(h * E_ + k) * HD_ + d]);
}

// ---- Pack Wo^T: B element (k,n) = Wo[n][k]; K=N=1024 ----
__global__ void pack_wot(const float* __restrict__ Wo, unsigned short* __restrict__ dst) {
  int idx  = blockIdx.x * blockDim.x + threadIdx.x;
  int tile = idx >> 9, e = idx & 511;
  int nt = tile >> 5, kt = tile & 31;
  int lane = e >> 4, p = e & 15, j = p >> 1, half = p & 1;
  int nn = lane & 15, khi = lane >> 4;
  int kk = (khi << 4) + 2 * j + half;
  int n = nt * 16 + nn, k = kt * 32 + kk;
  dst[idx] = f2bf(Wo[n * E_ + k]);
}

__global__ void pack_bias(const float* __restrict__ bq, const float* __restrict__ bk,
                          const float* __restrict__ bv, float* __restrict__ dst) {
  int n = blockIdx.x * blockDim.x + threadIdx.x;
  int which = n >> 10, e = n & 1023;
  dst[n] = (which == 0) ? bq[e] : ((which == 1) ? bk[e] : bv[e]);
}

// ---- Tiled GEMM: wave tile 32x64 (2Mx4N tiles, 8 wmma / 12 loads per k-step) ----
// WG = 8 waves in 2(M)x4(N) => WG tile 64x256.
__global__ void gemm_bf16(const uint4* __restrict__ Asw, const uint4* __restrict__ Bsw,
                          const float* __restrict__ bias,
                          unsigned short* __restrict__ Qd, unsigned short* __restrict__ Kd,
                          unsigned short* __restrict__ Vd, float* __restrict__ Fd,
                          int N, int KT, int mode) {
  int tid = threadIdx.x, lane = tid & 31, w = tid >> 5;
  int wm = w & 1, wn = w >> 1;
  int mt0 = blockIdx.y * 4 + wm * 2;
  int nt0 = blockIdx.x * 16 + wn * 4;
  v8f c[2][4] = {};
  for (int kt = 0; kt < KT; ++kt) {
    Frag a[2], b[4];
    #pragma unroll
    for (int i = 0; i < 2; ++i) {
      const uint4* pa = Asw + ((mt0 + i) * KT + kt) * 32 + lane * 2;
      a[i].q[0] = pa[0]; a[i].q[1] = pa[1];
    }
    #pragma unroll
    for (int jj = 0; jj < 4; ++jj) {
      const uint4* pb = Bsw + ((nt0 + jj) * KT + kt) * 32 + lane * 2;
      b[jj].q[0] = pb[0]; b[jj].q[1] = pb[1];
    }
    #pragma unroll
    for (int i = 0; i < 2; ++i)
      #pragma unroll
      for (int jj = 0; jj < 4; ++jj)
        c[i][jj] = __builtin_amdgcn_wmma_f32_16x16x32_bf16(
            false, a[i].v, false, b[jj].v, (short)0, c[i][jj], false, false);
  }
  int nlocal = lane & 15, mhalf = lane >> 4;
  #pragma unroll
  for (int i = 0; i < 2; ++i)
    #pragma unroll
    for (int jj = 0; jj < 4; ++jj) {
      int gm0 = (mt0 + i) * 16, gn0 = (nt0 + jj) * 16;
      int n = gn0 + nlocal;
      float bb = bias[n];
      #pragma unroll
      for (int j = 0; j < 8; ++j) {
        int m = gm0 + mhalf * 8 + j;
        float v = c[i][jj][j] + bb;
        if (mode == 0) {
          int which = n >> 10, ecol = n & 1023;
          int h = ecol >> 6, d = ecol & 63;
          int b2 = m >> 10, s = m & 1023;
          int bh = b2 * H_ + h;
          if (which == 0) {                   // Q: A-layout
            int tile = bh * 128 + (s >> 4) * 2 + (d >> 5);
            Qd[tile * 512 + a_elem(s & 15, d & 31)] = f2bf(v);
          } else if (which == 1) {            // K: B-layout (N=s, K=d)
            int tile = bh * 128 + (s >> 4) * 2 + (d >> 5);
            Kd[tile * 512 + b_elem(d & 31, s & 15)] = f2bf(v);
          } else {                            // V: B-layout (N=d, K=s)
            int tile = bh * 128 + (d >> 4) * 32 + (s >> 5);
            Vd[tile * 512 + b_elem(s & 31, d & 15)] = f2bf(v);
          }
        } else {
          Fd[m * N + n] = v;
        }
      }
    }
}

// ---- Flash attention with TDM double-buffered K/V staging in LDS ----
// 8 waves/WG share one (b,h); wave w handles q block qb = (bx&7)*8 + w.
__global__ void attn_kernel(const uint4* __restrict__ Qsw,
                            const unsigned short* __restrict__ Kd,
                            const unsigned short* __restrict__ Vd,
                            unsigned short* __restrict__ Att) {
  __shared__ __align__(16) unsigned short Pb[8][512];   // per-wave 16x32 bf16 P tile
  __shared__ __align__(16) uint4 KT_[2][256];           // 4KB: 32 keys x 64d (4 B-tiles)
  __shared__ __align__(16) uint4 VT_[2][256];           // 4KB: 4 B-tiles (n=0..3) for kt=it
  int tid = threadIdx.x, lane = tid & 31, w = tid >> 5;
  int bh = blockIdx.x >> 3;
  int qb = (blockIdx.x & 7) * 8 + w;

  const uint4* Qb = Qsw + (bh * 128 + qb * 2) * 32;
  unsigned long long kbase = (unsigned long long)(uintptr_t)Kd + ((unsigned long long)bh << 17);
  unsigned long long vbase = (unsigned long long)(uintptr_t)Vd + ((unsigned long long)bh << 17);

  Frag a0, a1;                                          // Q rows, d=0..31 / 32..63
  a0.q[0] = Qb[lane * 2];      a0.q[1] = Qb[lane * 2 + 1];
  a1.q[0] = Qb[32 + lane * 2]; a1.q[1] = Qb[32 + lane * 2 + 1];

  if (w == 0) {                                         // prologue: stage it=0
    tdm_load_1d((unsigned)(uintptr_t)&KT_[0][0], kbase, 512);
    tdm_load_2d((unsigned)(uintptr_t)&VT_[0][0], vbase, 128, 4, 4096);
  }

  float mrow[8], lrow[8];
  v8f o[4] = {};
  #pragma unroll
  for (int j = 0; j < 8; ++j) { mrow[j] = -1e30f; lrow[j] = 0.f; }
  int mhalf = lane >> 4, col = lane & 15;

  for (int it = 0; it < 32; ++it) {
    int cur = it & 1, nxt = cur ^ 1;
    if (w == 0) {
      if (it + 1 < 32) {                                // prefetch next chunk
        tdm_load_1d((unsigned)(uintptr_t)&KT_[nxt][0], kbase + ((unsigned long long)(it + 1) << 12), 512);
        tdm_load_2d((unsigned)(uintptr_t)&VT_[nxt][0], vbase + ((unsigned long long)(it + 1) << 10), 128, 4, 4096);
        __builtin_amdgcn_s_wait_tensorcnt(2);           // current chunk complete
      } else {
        __builtin_amdgcn_s_wait_tensorcnt(0);
      }
    }
    __syncthreads();                                    // staged data visible to all waves

    Frag bk00, bk01, bk10, bk11;
    const uint4* kp = &KT_[cur][0];
    bk00.q[0] = kp[  0 + lane * 2]; bk00.q[1] = kp[  1 + lane * 2];
    bk01.q[0] = kp[ 64 + lane * 2]; bk01.q[1] = kp[ 65 + lane * 2];
    bk10.q[0] = kp[128 + lane * 2]; bk10.q[1] = kp[129 + lane * 2];
    bk11.q[0] = kp[192 + lane * 2]; bk11.q[1] = kp[193 + lane * 2];

    v8f s0 = {}, s1 = {};
    s0 = __builtin_amdgcn_wmma_f32_16x16x32_bf16(false, a0.v, false, bk00.v, (short)0, s0, false, false);
    s0 = __builtin_amdgcn_wmma_f32_16x16x32_bf16(false, a1.v, false, bk01.v, (short)0, s0, false, false);
    s1 = __builtin_amdgcn_wmma_f32_16x16x32_bf16(false, a0.v, false, bk10.v, (short)0, s1, false, false);
    s1 = __builtin_amdgcn_wmma_f32_16x16x32_bf16(false, a1.v, false, bk11.v, (short)0, s1, false, false);

    float al[8];
    #pragma unroll
    for (int j = 0; j < 8; ++j) {
      float x0 = s0[j] * 0.125f, x1 = s1[j] * 0.125f;   // 1/sqrt(HD)
      float mx = fmaxf(x0, x1);
      mx = fmaxf(mx, __shfl_xor(mx, 1, 32));
      mx = fmaxf(mx, __shfl_xor(mx, 2, 32));
      mx = fmaxf(mx, __shfl_xor(mx, 4, 32));
      mx = fmaxf(mx, __shfl_xor(mx, 8, 32));
      float mnew  = fmaxf(mrow[j], mx);
      float alpha = __expf(mrow[j] - mnew);
      float p0v = __expf(x0 - mnew), p1v = __expf(x1 - mnew);
      float rs = p0v + p1v;
      rs += __shfl_xor(rs, 1, 32);
      rs += __shfl_xor(rs, 2, 32);
      rs += __shfl_xor(rs, 4, 32);
      rs += __shfl_xor(rs, 8, 32);
      lrow[j] = lrow[j] * alpha + rs;
      mrow[j] = mnew;
      al[j]   = alpha;
      int mr = mhalf * 8 + j;
      Pb[w][a_elem(mr, col)]      = f2bf(p0v);          // P -> A-fragment layout in LDS
      Pb[w][a_elem(mr, 16 + col)] = f2bf(p1v);
    }
    #pragma unroll
    for (int n = 0; n < 4; ++n)
      #pragma unroll
      for (int j = 0; j < 8; ++j) o[n][j] *= al[j];

    asm volatile("s_wait_dscnt 0" ::: "memory");        // cross-lane P hand-off

    Frag ap;
    const uint4* pp = (const uint4*)&Pb[w][0];
    ap.q[0] = pp[lane * 2]; ap.q[1] = pp[lane * 2 + 1];
    #pragma unroll
    for (int n = 0; n < 4; ++n) {
      Frag bv2;
      const uint4* pv = &VT_[cur][n * 64 + lane * 2];
      bv2.q[0] = pv[0]; bv2.q[1] = pv[1];
      o[n] = __builtin_amdgcn_wmma_f32_16x16x32_bf16(false, ap.v, false, bv2.v, (short)0, o[n], false, false);
    }
    __syncthreads();                                    // all reads done before buffer reuse
  }

  int b2 = bh >> 4, h = bh & 15;
  #pragma unroll
  for (int j = 0; j < 8; ++j) {
    float inv = 1.f / lrow[j];
    int m = b2 * S_ + qb * 16 + mhalf * 8 + j;
    #pragma unroll
    for (int n = 0; n < 4; ++n) {
      int cc   = h * 64 + n * 16 + col;
      int tile = (m >> 4) * 32 + (cc >> 5);
      Att[tile * 512 + a_elem(m & 15, cc & 31)] = f2bf(o[n][j] * inv);
    }
  }
}

extern "C" void kernel_launch(void* const* d_in, const int* in_sizes, int n_in,
                              void* d_out, int out_size, void* d_ws, size_t ws_size,
                              hipStream_t stream) {
  const float* x  = (const float*)d_in[0];
  const float* Wq = (const float*)d_in[2];
  const float* bq = (const float*)d_in[3];
  const float* Wk = (const float*)d_in[4];
  const float* bk = (const float*)d_in[5];
  const float* Wv = (const float*)d_in[6];
  const float* bv = (const float*)d_in[7];
  const float* Wo = (const float*)d_in[8];
  const float* bo = (const float*)d_in[9];

  char* ws = (char*)d_ws;
  unsigned short* Xb   = (unsigned short*)(ws);             // 8 MB
  unsigned short* Wqkv = (unsigned short*)(ws + 8388608);   // 6 MB
  unsigned short* WoT  = (unsigned short*)(ws + 14680064);  // 2 MB
  float*          qb_  = (float*)        (ws + 16777216);   // 12 KB
  unsigned short* Qd   = (unsigned short*)(ws + 16842752);  // 8 MB
  unsigned short* Kd   = (unsigned short*)(ws + 25231360);  // 8 MB
  unsigned short* Vd   = (unsigned short*)(ws + 33619968);  // 8 MB
  unsigned short* Att  = (unsigned short*)(ws + 42008576);  // 8 MB

  pack_a_f32<<<16384, 256, 0, stream>>>(x, Xb, 32);
  pack_wqkv <<<12288, 256, 0, stream>>>(Wq, Wk, Wv, Wqkv);
  pack_wot  <<< 4096, 256, 0, stream>>>(Wo, WoT);
  pack_bias <<<   12, 256, 0, stream>>>(bq, bk, bv, qb_);

  gemm_bf16<<<dim3(12, 64), 256, 0, stream>>>((const uint4*)Xb, (const uint4*)Wqkv, qb_,
                                              Qd, Kd, Vd, nullptr, 3072, 32, 0);
  attn_kernel<<<512, 256, 0, stream>>>((const uint4*)Qd, Kd, Vd, Att);
  gemm_bf16<<<dim3(4, 64), 256, 0, stream>>>((const uint4*)Att, (const uint4*)WoT, bo,
                                             nullptr, nullptr, nullptr, (float*)d_out, 1024, 32, 1);
}
